// MultiHeadAttention_17712445129023
// MI455X (gfx1250) — matrix operand order repaired and checked by reading, not compiled, not run on hardware
//
#include <hip/hip_runtime.h>
#include <hip/hip_bf16.h>

typedef __bf16 v16bf __attribute__((ext_vector_type(16)));
typedef __bf16 v8bf  __attribute__((ext_vector_type(8)));
typedef float  v8f   __attribute__((ext_vector_type(8)));

// ---------------------------------------------------------------------------
// Fragment builders for V_WMMA_F32_16X16X32_BF16 (wave32).
//  A (16x32, MxK): lane = half*16+m holds row m; VGPR0-3 = K[base+half*8 .. +7],
//                  VGPR4-7 = K[base+16+half*8 .. +7]  (ISA 7.12.2, 16-bit A).
//  B (32x16, KxN): lane = half*16+n holds col n; same K order as A
//                  -> we keep B transposed [n][k] in LDS.
// ---------------------------------------------------------------------------
__device__ __forceinline__ v16bf frag_cat(const __bf16* p0, const __bf16* p1) {
  v8bf lo = *(const v8bf*)p0;
  v8bf hi = *(const v8bf*)p1;
  return __builtin_shufflevector(lo, hi, 0,1,2,3,4,5,6,7,8,9,10,11,12,13,14,15);
}

__device__ __forceinline__ v16bf a_frag(const __bf16* tile, int ld, int row0, int kbase) {
  int lane = threadIdx.x & 31;
  int half = lane >> 4, m = lane & 15;
  const __bf16* r = tile + (row0 + m) * ld + kbase + half * 8;
  return frag_cat(r, r + 16);
}

__device__ __forceinline__ v16bf b_frag(const __bf16* bt, int ld, int n0, int kbase) {
  int lane = threadIdx.x & 31;
  int half = lane >> 4, n = lane & 15;
  const __bf16* r = bt + (n0 + n) * ld + kbase + half * 8;
  return frag_cat(r, r + 16);
}

__device__ __forceinline__ v8f wmma_bf16(v16bf a, v16bf b, v8f c) {
  return __builtin_amdgcn_wmma_f32_16x16x32_bf16(false, a, false, b, (short)0, c,
                                                 false, false);
}

// ---------------------------------------------------------------------------
// Conversion / packing kernels (fp32 -> bf16, weight relayout)
// ---------------------------------------------------------------------------
__global__ void k_cvt(const float* __restrict__ src, __bf16* __restrict__ dst, int n) {
  int i = blockIdx.x * blockDim.x + threadIdx.x;
  if (i < n) dst[i] = (__bf16)src[i];
}

// W[h][c][d] (16,1024,64) -> Wp[c][h*64+d]  (K x N GEMM operand)
__global__ void k_pack_w(const float* __restrict__ w, __bf16* __restrict__ wp) {
  int i = blockIdx.x * blockDim.x + threadIdx.x;   // over 1024*1024
  int c = i >> 10, n = i & 1023;
  int h = n >> 6, d = n & 63;
  wp[i] = (__bf16)w[(size_t)(h * 1024 + c) * 64 + d];
}

// Wo[n][k] -> WoT[k][n]
__global__ void k_pack_woT(const float* __restrict__ wo, __bf16* __restrict__ wt) {
  int i = blockIdx.x * blockDim.x + threadIdx.x;   // over 1024*1024
  int k = i >> 10, n = i & 1023;
  wt[i] = (__bf16)wo[(size_t)n * 1024 + k];
}

// ---------------------------------------------------------------------------
// bf16 GEMM:  C[M,N] = A[M,K] * B[K,N]   (row-major A and B)
// block = 256 threads (8 waves), tile M=128, N=64, K-step 64 (2 WMMA k-chunks).
// OUTF32: write float + bias (final projection); else write bf16.
// ---------------------------------------------------------------------------
template <bool OUTF32>
__global__ void k_gemm(const __bf16* __restrict__ A, const __bf16* __restrict__ Bm,
                       __bf16* __restrict__ Cb, float* __restrict__ Cf,
                       const float* __restrict__ bias, int M, int N, int K) {
  __shared__ __bf16 As[128 * 72];   // [row][k], stride 72 (16B aligned, bank-friendly)
  __shared__ __bf16 Bt[64 * 72];    // B transposed: [n][k]
  const int t  = threadIdx.x;
  const int w  = t >> 5;
  const int mw = w >> 1, nw = w & 1;            // waves: 4 x 2 over (M,N)
  const int m0 = blockIdx.y * 128, n0 = blockIdx.x * 64;

  v8f acc[2][2] = {};

  for (int k0 = 0; k0 < K; k0 += 64) {
    __syncthreads();
    {   // A tile 128x64: thread -> (row = t>>1, 32 elems at (t&1)*32)
      int row = t >> 1, koff = (t & 1) * 32;
      const uint4* g = (const uint4*)(A + (size_t)(m0 + row) * K + k0 + koff);
      uint4* s = (uint4*)(As + row * 72 + koff);
      s[0] = g[0]; s[1] = g[1]; s[2] = g[2]; s[3] = g[3];
      if (k0 + 64 < K) __builtin_prefetch(A + (size_t)(m0 + row) * K + k0 + 64 + koff, 0, 0);
    }
    {   // B tile 64x64 -> transposed LDS
      int kk = t >> 2, nb = (t & 3) * 16;
      const __bf16* g = Bm + (size_t)(k0 + kk) * N + n0 + nb;
      uint4 val0 = ((const uint4*)g)[0];
      uint4 val1 = ((const uint4*)g)[1];
      const __bf16* v0 = (const __bf16*)&val0;
      const __bf16* v1 = (const __bf16*)&val1;
#pragma unroll
      for (int j = 0; j < 8; ++j) Bt[(nb + j) * 72 + kk] = v0[j];
#pragma unroll
      for (int j = 0; j < 8; ++j) Bt[(nb + 8 + j) * 72 + kk] = v1[j];
      if (k0 + 64 < K) __builtin_prefetch(Bm + (size_t)(k0 + 64 + kk) * N + n0 + nb, 0, 0);
    }
    __syncthreads();

#pragma unroll
    for (int kc = 0; kc < 64; kc += 32) {
      v16bf a0 = a_frag(As, 72, mw * 32,      kc);
      v16bf a1 = a_frag(As, 72, mw * 32 + 16, kc);
      v16bf b0 = b_frag(Bt, 72, nw * 32,      kc);
      v16bf b1 = b_frag(Bt, 72, nw * 32 + 16, kc);
      acc[0][0] = wmma_bf16(a0, b0, acc[0][0]);
      acc[0][1] = wmma_bf16(a0, b1, acc[0][1]);
      acc[1][0] = wmma_bf16(a1, b0, acc[1][0]);
      acc[1][1] = wmma_bf16(a1, b1, acc[1][1]);
    }
  }

  const int lane = t & 31, half = lane >> 4, nn = lane & 15;
#pragma unroll
  for (int mt = 0; mt < 2; ++mt)
#pragma unroll
    for (int nt = 0; nt < 2; ++nt) {
      int col = n0 + nw * 32 + nt * 16 + nn;
      float bv = OUTF32 ? bias[col] : 0.f;
#pragma unroll
      for (int i = 0; i < 8; ++i) {
        int row = m0 + mw * 32 + mt * 16 + i + 8 * half;
        float vv = acc[mt][nt][i] + bv;
        if (OUTF32) Cf[(size_t)row * N + col] = vv;
        else        Cb[(size_t)row * N + col] = (__bf16)vv;
      }
    }
}

// ---------------------------------------------------------------------------
// Flash attention over one (b,h): block = 128 threads (4 waves).
// Each wave owns 16 query rows; key/value tiles of 64 iterate causally.
// q,k,v,out layout: [(b*T + t) * 1024 + h*64 + d]  (bf16)
// Q fragments are re-built from LDS every key-tile iteration to keep their
// live range tiny (round 1 showed the compiler spilling them to scratch).
// ---------------------------------------------------------------------------
__global__ void k_attn(const __bf16* __restrict__ qb, const __bf16* __restrict__ kb,
                       const __bf16* __restrict__ vb, __bf16* __restrict__ ob, int T) {
  __shared__ __bf16 Qt[64 * 72];        // [q][d]
  __shared__ __bf16 Kt[64 * 72];        // [key][d] == transposed-B for Q*K^T
  __shared__ __bf16 Vt[64 * 72];        // [d][key] == transposed-B for P*V
  __shared__ __bf16 Pt[4 * 16 * 72];    // per-wave P slab (D-layout -> A-layout)

  const int t = threadIdx.x;
  const int w = t >> 5;
  const int lane = t & 31, half = lane >> 4, nn = lane & 15;
  const int qtile = blockIdx.x;
  const int bh = blockIdx.y;
  const int b = bh >> 4, h = bh & 15;
  const int q0 = qtile * 64;
  const size_t base = (size_t)b * T * 1024 + (size_t)h * 64;

  {   // Q tile 64x64: thread -> (row=t>>1, 32 elems at (t&1)*32)
    int row = t >> 1, doff = (t & 1) * 32;
    const uint4* g = (const uint4*)(qb + base + (size_t)(q0 + row) * 1024 + doff);
    uint4* s = (uint4*)(Qt + row * 72 + doff);
    s[0] = g[0]; s[1] = g[1]; s[2] = g[2]; s[3] = g[3];
  }

  v8f o[4] = {};
  float mrow[8], lrow[8];
#pragma unroll
  for (int i = 0; i < 8; ++i) { mrow[i] = -3.0e38f; lrow[i] = 0.f; }

  const int nkt = qtile + 1;                   // causal: keys up to this q-block
  for (int kt = 0; kt < nkt; ++kt) {
    const int kt0 = kt * 64;
    __syncthreads();                           // prior tile fully consumed / Q ready
    {
      int row = t >> 1, doff = (t & 1) * 32;
      const uint4* g = (const uint4*)(kb + base + (size_t)(kt0 + row) * 1024 + doff);
      uint4* s = (uint4*)(Kt + row * 72 + doff);
      s[0] = g[0]; s[1] = g[1]; s[2] = g[2]; s[3] = g[3];
      // V transpose: two 16-element chunks to keep staging registers small
      const uint4* gv = (const uint4*)(vb + base + (size_t)(kt0 + row) * 1024 + doff);
#pragma unroll
      for (int c = 0; c < 2; ++c) {
        uint4 t0 = gv[2 * c], t1 = gv[2 * c + 1];
        const __bf16* v0 = (const __bf16*)&t0;
        const __bf16* v1 = (const __bf16*)&t1;
#pragma unroll
        for (int j = 0; j < 8; ++j) Vt[(doff + c * 16 + j) * 72 + row] = v0[j];
#pragma unroll
        for (int j = 0; j < 8; ++j) Vt[(doff + c * 16 + 8 + j) * 72 + row] = v1[j];
      }
    }
    __syncthreads();

    // S = Q K^T  (16 rows x 64 keys per wave); Q frags rebuilt from LDS here
    v8f s4[4];
    {
      v16bf qa0 = a_frag(Qt, 72, w * 16, 0);
      v16bf qa1 = a_frag(Qt, 72, w * 16, 32);
#pragma unroll
      for (int j = 0; j < 4; ++j) {
        v8f s = {};
        s = wmma_bf16(qa0, b_frag(Kt, 72, j * 16, 0),  s);
        s = wmma_bf16(qa1, b_frag(Kt, 72, j * 16, 32), s);
        s4[j] = s;
      }
    }

    // scale (1/sqrt(1024)), causal mask, online softmax -- single pass per row
#pragma unroll
    for (int i = 0; i < 8; ++i) {
      int qidx = q0 + w * 16 + i + 8 * half;
      float mx = -3.0e38f;
#pragma unroll
      for (int j = 0; j < 4; ++j) {
        int kidx = kt0 + j * 16 + nn;
        float sv = s4[j][i] * 0.03125f;
        sv = (kidx > qidx) ? -1.0e30f : sv;
        s4[j][i] = sv;
        mx = fmaxf(mx, sv);
      }
#pragma unroll
      for (int off = 1; off < 16; off <<= 1)          // stays within lane half
        mx = fmaxf(mx, __shfl_xor(mx, off, 32));
      float nm = fmaxf(mrow[i], mx);
      float alpha = __expf(mrow[i] - nm);
      float rs = 0.f;
#pragma unroll
      for (int j = 0; j < 4; ++j) {
        float p = __expf(s4[j][i] - nm);
        s4[j][i] = p;
        rs += p;
      }
#pragma unroll
      for (int off = 1; off < 16; off <<= 1)
        rs += __shfl_xor(rs, off, 32);
      lrow[i] = lrow[i] * alpha + rs;
      mrow[i] = nm;
#pragma unroll
      for (int j = 0; j < 4; ++j) o[j][i] *= alpha;
    }

    // P: D-layout regs -> per-wave LDS slab -> A-layout frags (wave-local,
    // LDS ops from one wave complete in order; barrier only for the compiler)
    __bf16* Pw = Pt + w * (16 * 72);
#pragma unroll
    for (int j = 0; j < 4; ++j)
#pragma unroll
      for (int i = 0; i < 8; ++i)
        Pw[(i + 8 * half) * 72 + j * 16 + nn] = (__bf16)s4[j][i];
    __builtin_amdgcn_wave_barrier();
    {
      v16bf pa0 = a_frag(Pw, 72, 0, 0);
      v16bf pa1 = a_frag(Pw, 72, 0, 32);
      // O += P * V
#pragma unroll
      for (int j = 0; j < 4; ++j) {
        o[j] = wmma_bf16(pa0, b_frag(Vt, 72, j * 16, 0),  o[j]);
        o[j] = wmma_bf16(pa1, b_frag(Vt, 72, j * 16, 32), o[j]);
      }
    }
  }

  // finalize: O /= l, write concat-head bf16
#pragma unroll
  for (int j = 0; j < 4; ++j)
#pragma unroll
    for (int i = 0; i < 8; ++i) {
      int qidx = q0 + w * 16 + i + 8 * half;
      int d = j * 16 + nn;
      ob[base + (size_t)qidx * 1024 + d] = (__bf16)(o[j][i] / lrow[i]);
    }
}

// ---------------------------------------------------------------------------
extern "C" void kernel_launch(void* const* d_in, const int* in_sizes, int n_in,
                              void* d_out, int out_size, void* d_ws, size_t ws_size,
                              hipStream_t stream) {
  (void)in_sizes; (void)n_in; (void)out_size; (void)ws_size;
  const float* x  = (const float*)d_in[0];
  const float* Wq = (const float*)d_in[1];
  const float* Wk = (const float*)d_in[2];
  const float* Wv = (const float*)d_in[3];
  const float* Wo = (const float*)d_in[4];
  const float* bo = (const float*)d_in[5];
  float* out = (float*)d_out;

  const int B = 2, T = 2048, C = 1024;
  const int M = B * T;                           // 4096

  char* ws = (char*)d_ws;
  size_t off = 0;
  auto alloc = [&](size_t bytes) -> void* {
    void* p = ws + off;
    off += (bytes + 255) & ~(size_t)255;
    return p;
  };
  __bf16* xbf  = (__bf16*)alloc((size_t)M * C * 2);   // x in bf16
  __bf16* Wqp  = (__bf16*)alloc((size_t)C * C * 2);
  __bf16* Wkp  = (__bf16*)alloc((size_t)C * C * 2);
  __bf16* Wvp  = (__bf16*)alloc((size_t)C * C * 2);
  __bf16* WoT  = (__bf16*)alloc((size_t)C * C * 2);
  __bf16* qbuf = (__bf16*)alloc((size_t)M * C * 2);
  __bf16* kbuf = (__bf16*)alloc((size_t)M * C * 2);
  __bf16* vbuf = (__bf16*)alloc((size_t)M * C * 2);
  __bf16* abuf = (__bf16*)alloc((size_t)M * C * 2);   // attention out (concat heads)

  k_cvt<<<(M * C) / 256, 256, 0, stream>>>(x, xbf, M * C);
  k_pack_w<<<(C * C) / 256, 256, 0, stream>>>(Wq, Wqp);
  k_pack_w<<<(C * C) / 256, 256, 0, stream>>>(Wk, Wkp);
  k_pack_w<<<(C * C) / 256, 256, 0, stream>>>(Wv, Wvp);
  k_pack_woT<<<(C * C) / 256, 256, 0, stream>>>(Wo, WoT);

  dim3 gg(C / 64, M / 128);                      // (16, 32)
  k_gemm<false><<<gg, 256, 0, stream>>>(xbf, Wqp, qbuf, nullptr, nullptr, M, C, C);
  k_gemm<false><<<gg, 256, 0, stream>>>(xbf, Wkp, kbuf, nullptr, nullptr, M, C, C);
  k_gemm<false><<<gg, 256, 0, stream>>>(xbf, Wvp, vbuf, nullptr, nullptr, M, C, C);

  dim3 ga(T / 64, B * 16);                       // (32 q-tiles, 32 b*h)
  k_attn<<<ga, 128, 0, stream>>>(qbuf, kbuf, vbuf, abuf, T);

  k_gemm<true><<<gg, 256, 0, stream>>>(abuf, WoT, nullptr, out, bo, M, C, C);
}
